// PTF_Model_13700945674489
// MI455X (gfx1250) — matrix operand order repaired and checked by reading, not compiled, not run on hardware
//
#include <hip/hip_runtime.h>

// ---------------------------------------------------------------------------
// HCNN partial-teacher-forcing recurrence for MI455X (gfx1250, wave32, WMMA).
//
//   s_mod = [y_t, s[:,32:]] ;  s' = tanh(s_mod @ A^T)
//
// Batch rows independent -> 8 blocks x 16-row tiles (WMMA M=16). A converted
// once to bf16 in d_ws. Per-step B-operand traffic split: K in [0,256) cached
// in LDS (fragment-major, conflict-free, loaded once), K in [256,512) streamed
// from L2. kt-outer / nt-inner loop reuses each A-frag across 4 WMMAs with 4
// live accumulators. Native v_tanh_f32; all 638 MB of outputs via coalesced
// nontemporal stores during the staging phase.
// ---------------------------------------------------------------------------

typedef __bf16 bf16;
typedef __attribute__((ext_vector_type(16))) __bf16 v16bf;
typedef __attribute__((ext_vector_type(8)))  __bf16 v8bf;
typedef __attribute__((ext_vector_type(4)))  __bf16 v4bf;
typedef __attribute__((ext_vector_type(8)))  float  v8f;
typedef __attribute__((ext_vector_type(4)))  float  v4f;

#define BB    128
#define LL    2048
#define NOBS  32
#define NHID  512
#define MT    16             // batch rows per block (WMMA M)
#define SMS   (NHID + 8)     // padded bf16 row stride for smod (1040 B)
#define KC    256            // K range of B-operand cached in LDS
#define NKT   (KC / 32)      // cached k-tiles (8)
#define NTILE (NHID / 16)    // 32 N-tiles total

static __device__ __forceinline__ float fast_tanh(float x) {
#if __has_builtin(__builtin_amdgcn_tanhf)
    return __builtin_amdgcn_tanhf(x);     // native V_TANH_F32 (CDNA5 TRANS op)
#else
    return tanhf(x);
#endif
}

__global__ void hcnn_convert_A(const float* __restrict__ A,
                               bf16* __restrict__ Abf, int n) {
    int i = blockIdx.x * blockDim.x + threadIdx.x;
    if (i < n) Abf[i] = (bf16)A[i];
}

__global__ __launch_bounds__(256)
void hcnn_recurrence(const float* __restrict__ data,   // [B, L, 32]
                     const float* __restrict__ h0,     // [512]
                     const bf16*  __restrict__ Abf,    // [512, 512] row-major A[n][k]
                     float* __restrict__ exps,         // [B, L, 32]
                     float* __restrict__ states,       // [B, L, 512]
                     float* __restrict__ deltas,       // [B, L, 32]
                     float* __restrict__ partials)     // [B, L, 32]
{
    __shared__ float s_cur[MT][NHID];                      // 32 KB
    __shared__ bf16  smod[MT][SMS];                        // ~16.6 KB
    // Fragment-major B cache: [tile][ktile][lane] x 16 bf16 (32 B chunk per
    // lane) -> each wave's B-frag load is 1 KB contiguous, conflict-free.
    __shared__ __attribute__((aligned(32))) bf16 Bc[NTILE * NKT * 32 * 16]; // 256 KB

    const int row0 = blockIdx.x * MT;   // 8 blocks cover B=128
    const int tid  = threadIdx.x;
    const int lane = tid & 31;
    const int wave = tid >> 5;

    // WMMA lane decomposition (ISA 16-bit A/B/D layouts, wave32)
    const int m16   = lane & 15;        // A-frag row / B,D-frag column index
    const int khalf = lane >> 4;        // K/M half selector

    // ---- init s0 = broadcast(h0) ----
    for (int i = tid; i < MT * NHID; i += 256) {
        int r = i >> 9, c = i & (NHID - 1);
        s_cur[r][c] = h0[c];
    }

    // ---- prefill LDS B-cache (K in [0,KC)), fragment-major ----
    for (int ch = tid; ch < NTILE * NKT * 32; ch += 256) {
        int ln   = ch & 31;
        int kt32 = (ch >> 5) % NKT;
        int tile = (ch >> 5) / NKT;
        int col  = tile * 16 + (ln & 15);
        int k    = kt32 * 32 + (ln >> 4) * 16;
        v16bf v = *(const v16bf*)&Abf[(size_t)col * NHID + k];
        *(v16bf*)&Bc[(size_t)ch * 16] = v;
    }
    __syncthreads();

    for (int t = 0; t < LL - 1; ++t) {
        // ---- stage bf16 s_mod; emit states[:,t], y_hat, delta, partial ----
        for (int i = tid * 4; i < MT * NHID; i += 256 * 4) {
            int r = i >> 9, c = i & (NHID - 1);
            v4f sv = *(const v4f*)&s_cur[r][c];

            size_t so = ((size_t)(row0 + r) * LL + t) * NHID + c;
            __builtin_nontemporal_store(sv, (v4f*)&states[so]);

            v4f mv = sv;
            if (c < NOBS) {  // whole float4 is inside the observation window
                size_t o = ((size_t)(row0 + r) * LL + t) * NOBS + c;
                v4f y = *(const v4f*)&data[o];
                v4f d = y - sv;
                __builtin_nontemporal_store(sv, (v4f*)&exps[o]);
                __builtin_nontemporal_store(d,  (v4f*)&deltas[o]);
                __builtin_nontemporal_store(d,  (v4f*)&partials[o]);
                mv = y;      // first 32 cols of s_mod are the observation
            }
            v4bf p = { (bf16)mv[0], (bf16)mv[1], (bf16)mv[2], (bf16)mv[3] };
            *(v4bf*)&smod[r][c] = p;
        }
        __syncthreads();

        // ---- s' = tanh(s_mod @ A^T): wave owns 4 N-tiles (64 columns) ----
        const int ncolbase = wave * 64;
        v8f z = {0.f, 0.f, 0.f, 0.f, 0.f, 0.f, 0.f, 0.f};
        v8f acc[4] = {z, z, z, z};

        // Phase 1: cached K from LDS (conflict-free fragment-major reads)
#pragma unroll 2
        for (int kt = 0; kt < KC; kt += 32) {
            const v8bf a0 = *(const v8bf*)&smod[m16][kt + khalf * 16];
            const v8bf a1 = *(const v8bf*)&smod[m16][kt + khalf * 16 + 8];
            v16bf afrag = __builtin_shufflevector(
                a0, a1, 0, 1, 2, 3, 4, 5, 6, 7, 8, 9, 10, 11, 12, 13, 14, 15);
            const int kt32 = kt >> 5;
#pragma unroll
            for (int nt = 0; nt < 4; ++nt) {
                const int tile = wave * 4 + nt;
                const v16bf bfrag =
                    *(const v16bf*)&Bc[(size_t)(((tile * NKT) + kt32) * 32 + lane) * 16];
                acc[nt] = __builtin_amdgcn_wmma_f32_16x16x32_bf16(
                    false, afrag, false, bfrag, (short)0, acc[nt], false, false);
            }
        }

        // Phase 2: streamed K from L2 (A is 512 KB, L2-resident)
#pragma unroll 2
        for (int kt = KC; kt < NHID; kt += 32) {
            const v8bf a0 = *(const v8bf*)&smod[m16][kt + khalf * 16];
            const v8bf a1 = *(const v8bf*)&smod[m16][kt + khalf * 16 + 8];
            v16bf afrag = __builtin_shufflevector(
                a0, a1, 0, 1, 2, 3, 4, 5, 6, 7, 8, 9, 10, 11, 12, 13, 14, 15);
#pragma unroll
            for (int nt = 0; nt < 4; ++nt) {
                const int ccol = ncolbase + nt * 16 + m16;
                const v16bf bfrag =
                    *(const v16bf*)&Abf[(size_t)ccol * NHID + kt + khalf * 16];
                acc[nt] = __builtin_amdgcn_wmma_f32_16x16x32_bf16(
                    false, afrag, false, bfrag, (short)0, acc[nt], false, false);
            }
        }

        // ---- writeback: D vgpr j -> M = j + 8*khalf, N = m16 ----
#pragma unroll
        for (int nt = 0; nt < 4; ++nt) {
            const int ccol = ncolbase + nt * 16 + m16;
#pragma unroll
            for (int j = 0; j < 8; ++j)
                s_cur[j + 8 * khalf][ccol] = fast_tanh(acc[nt][j]);
        }
        __syncthreads();
    }

    // ---- epilogue: states[:,L-1] + last y_hat/delta (uses data[:, L-2]) ----
    for (int i = tid * 4; i < MT * NHID; i += 256 * 4) {
        int r = i >> 9, c = i & (NHID - 1);
        v4f sv = *(const v4f*)&s_cur[r][c];

        size_t so = ((size_t)(row0 + r) * LL + (LL - 1)) * NHID + c;
        __builtin_nontemporal_store(sv, (v4f*)&states[so]);

        if (c < NOBS) {
            size_t o  = ((size_t)(row0 + r) * LL + (LL - 1)) * NOBS + c;
            size_t oy = ((size_t)(row0 + r) * LL + (LL - 2)) * NOBS + c;
            v4f y = *(const v4f*)&data[oy];
            v4f d = y - sv;
            __builtin_nontemporal_store(sv, (v4f*)&exps[o]);
            __builtin_nontemporal_store(d,  (v4f*)&deltas[o]);
            __builtin_nontemporal_store(d,  (v4f*)&partials[o]);
        }
    }
}

extern "C" void kernel_launch(void* const* d_in, const int* in_sizes, int n_in,
                              void* d_out, int out_size, void* d_ws, size_t ws_size,
                              hipStream_t stream) {
    const float* data = (const float*)d_in[0];   // [128, 2048, 32]
    const float* A    = (const float*)d_in[1];   // [512, 512]
    const float* h0   = (const float*)d_in[2];   // [1, 512]
    // d_in[3] = prob (0) -> ptf dropout is identity; unused.

    float* out      = (float*)d_out;
    float* exps     = out;                                      // B*L*32
    float* states   = exps   + (size_t)BB * LL * NOBS;          // B*L*512
    float* deltas   = states + (size_t)BB * LL * NHID;          // B*L*32
    float* partials = deltas + (size_t)BB * LL * NOBS;          // B*L*32

    bf16* Abf = (bf16*)d_ws;                                    // 512 KB scratch

    const int nA = NHID * NHID;
    hcnn_convert_A<<<(nA + 255) / 256, 256, 0, stream>>>(A, Abf, nA);

    hcnn_recurrence<<<BB / MT, 256, 0, stream>>>(data, h0, Abf,
                                                 exps, states, deltas, partials);
}
